// PN2Net_2860448219405
// MI455X (gfx1250) — matrix operand order repaired and checked by reading, not compiled, not run on hardware
//
#include <hip/hip_runtime.h>
#include <math.h>

// ---------------------------------------------------------------------------
// Types for CDNA5 WMMA
// ---------------------------------------------------------------------------
typedef _Float16 f16_t;
typedef __attribute__((ext_vector_type(16))) _Float16 v16h;
typedef __attribute__((ext_vector_type(8)))  _Float16 v8h;
typedef __attribute__((ext_vector_type(8)))  float    v8f;

#define BATCH 8
#define NPTS  4096
#define KNBR  64

// ---------------------------------------------------------------------------
// BN fold: bias = b, a = g*rsqrt(var+eps), c = bt - mu*a
// ---------------------------------------------------------------------------
__global__ void pn2_bnprep(const float* __restrict__ b, const float* __restrict__ g,
                           const float* __restrict__ bt, const float* __restrict__ mu,
                           const float* __restrict__ var, int N,
                           float* __restrict__ bias, float* __restrict__ sa,
                           float* __restrict__ sc) {
    int i = blockIdx.x * blockDim.x + threadIdx.x;
    if (i >= N) return;
    float a = g[i] * rsqrtf(var[i] + 1e-5f);
    bias[i] = b[i];
    sa[i]   = a;
    sc[i]   = bt[i] - mu[i] * a;
}

// ---------------------------------------------------------------------------
// Pack weights f32 [cin, N] -> f16 WMMA-B fragment order [ntile][ktile][lane][16]
// B 32x16 f16 layout: lane n = lane&15, h = lane>>4, element i: k = (i<8) ? h*8 + i : 16 + h*8 + (i-8)
// ---------------------------------------------------------------------------
__global__ void pn2_packw(const float* __restrict__ W, int cin, int N, int Kp,
                          f16_t* __restrict__ out) {
    int kt = blockIdx.x, nt = blockIdx.y;
    int lane = threadIdx.x;                 // 32 threads
    int nn = nt * 16 + (lane & 15);
    int h  = lane >> 4;
    int ktiles = Kp >> 5;
    f16_t* o = out + (((size_t)nt * ktiles + kt) * 32 + lane) * 16;
#pragma unroll
    for (int i = 0; i < 16; i++) {
        int k = kt * 32 + ((i < 8) ? (h * 8 + i) : (16 + h * 8 + (i - 8)));
        float v = (k < cin) ? W[(size_t)k * N + nn] : 0.0f;
        o[i] = (f16_t)v;
    }
}

// ---------------------------------------------------------------------------
// Farthest point sampling: one block per batch. pos [B,n,3] -> idx [B,m]
// ---------------------------------------------------------------------------
__global__ void __launch_bounds__(512)
pn2_fps(const float* __restrict__ pos, int n, int m, int* __restrict__ out_idx) {
    __shared__ float mind[NPTS];
    __shared__ float rv[512];
    __shared__ int   ri[512];
    __shared__ int   s_last;
    int b = blockIdx.x;
    const float* P = pos + (size_t)b * n * 3;
    int* O = out_idx + (size_t)b * m;
    int t = threadIdx.x, T = blockDim.x;
    for (int j = t; j < n; j += T) mind[j] = 1e10f;
    if (t == 0) { O[0] = 0; s_last = 0; }
    __syncthreads();
    for (int s = 1; s < m; s++) {
        int last = s_last;
        float lx = P[last*3], ly = P[last*3+1], lz = P[last*3+2];
        float bv = -1.0f; int bi = 0;
        for (int j = t; j < n; j += T) {
            float dx = P[j*3] - lx, dy = P[j*3+1] - ly, dz = P[j*3+2] - lz;
            float d = dx*dx + dy*dy + dz*dz;
            float mn = mind[j]; mn = fminf(mn, d); mind[j] = mn;
            if (mn > bv) { bv = mn; bi = j; }
        }
        rv[t] = bv; ri[t] = bi;
        __syncthreads();
        for (int off = T >> 1; off > 0; off >>= 1) {
            if (t < off) {
                if (rv[t+off] > rv[t] || (rv[t+off] == rv[t] && ri[t+off] < ri[t])) {
                    rv[t] = rv[t+off]; ri[t] = ri[t+off];
                }
            }
            __syncthreads();
        }
        if (t == 0) { s_last = ri[0]; O[s] = ri[0]; }
        __syncthreads();
    }
}

// ---------------------------------------------------------------------------
// Gather sampled centers: ctr[b,i,:] = pos[b, idx[b,i], :]
// ---------------------------------------------------------------------------
__global__ void pn2_ctr(const float* __restrict__ pos, const int* __restrict__ idx,
                        float* __restrict__ ctr, int n, int m) {
    int t = blockIdx.x * blockDim.x + threadIdx.x;
    if (t >= BATCH * m) return;
    int b = t / m;
    int j = idx[t];
    const float* P = pos + ((size_t)b * n + j) * 3;
    float* C = ctr + (size_t)t * 3;
    C[0] = P[0]; C[1] = P[1]; C[2] = P[2];
}

// ---------------------------------------------------------------------------
// 64-NN per center with radius mask: one block per center, d2 row in LDS.
// ---------------------------------------------------------------------------
__global__ void __launch_bounds__(256)
pn2_knn(const float* __restrict__ pos, const float* __restrict__ ctr,
        int n, int m, float r2, int* __restrict__ nbr, unsigned char* __restrict__ mask) {
    __shared__ float d2[NPTS];
    __shared__ float rv[256];
    __shared__ int   ri[256];
    int c = blockIdx.x;            // [0, B*m)
    int b = c / m;
    const float* P = pos + (size_t)b * n * 3;
    float cx = ctr[(size_t)c*3], cy = ctr[(size_t)c*3+1], cz = ctr[(size_t)c*3+2];
    int t = threadIdx.x, T = blockDim.x;
    for (int j = t; j < n; j += T) {
        float dx = P[j*3] - cx, dy = P[j*3+1] - cy, dz = P[j*3+2] - cz;
        d2[j] = dx*dx + dy*dy + dz*dz;
    }
    __syncthreads();
    for (int k = 0; k < KNBR; k++) {
        float bv = 3.4e38f; int bi = 0;
        for (int j = t; j < n; j += T) {
            float v = d2[j];
            if (v < bv) { bv = v; bi = j; }
        }
        rv[t] = bv; ri[t] = bi;
        __syncthreads();
        for (int off = T >> 1; off > 0; off >>= 1) {
            if (t < off) {
                if (rv[t+off] < rv[t] || (rv[t+off] == rv[t] && ri[t+off] < ri[t])) {
                    rv[t] = rv[t+off]; ri[t] = ri[t+off];
                }
            }
            __syncthreads();
        }
        if (t == 0) {
            int s = ri[0];
            nbr[(size_t)c * KNBR + k]  = s;
            mask[(size_t)c * KNBR + k] = (rv[0] <= r2) ? 1 : 0;
            d2[s] = 3.39e38f;
        }
        __syncthreads();
    }
}

// ---------------------------------------------------------------------------
// Gather neighbor features + relative pos into f16, K-padded with zeros.
// out row r = (i,kn): [ feat[b,nbr,0..Cx) , pos[b,nbr]-ctr , 0-pad ] width Kp
// ---------------------------------------------------------------------------
__global__ void pn2_gather(const float* __restrict__ feat, const float* __restrict__ pos,
                           const float* __restrict__ ctr, const int* __restrict__ nbr,
                           int b, int n, int m, int Cx, int Kp, f16_t* __restrict__ out) {
    size_t R = (size_t)m * KNBR;
    size_t r = (size_t)blockIdx.x * blockDim.x + threadIdx.x;
    if (r >= R) return;
    size_t i = r >> 6;
    size_t c = (size_t)b * m + i;
    int j = nbr[c * KNBR + (r & 63)];
    const float* F  = feat + ((size_t)b * n + j) * Cx;
    const float* P  = pos  + ((size_t)b * n + j) * 3;
    const float* CT = ctr  + c * 3;
    f16_t* o = out + r * Kp;
    int q = 0;
    for (; q < Cx; q++) o[q] = (f16_t)F[q];
    o[q++] = (f16_t)(P[0] - CT[0]);
    o[q++] = (f16_t)(P[1] - CT[1]);
    o[q++] = (f16_t)(P[2] - CT[2]);
    for (; q < Kp; q++) o[q] = (f16_t)0.0f;
}

// SA3 input: [feat2[r,0..256), ctr2[r], pad] over all B*512 rows
__global__ void pn2_gather3(const float* __restrict__ feat, const float* __restrict__ ctr,
                            int Cx, int Kp, f16_t* __restrict__ out) {
    size_t R = (size_t)BATCH * 512;
    size_t r = (size_t)blockIdx.x * blockDim.x + threadIdx.x;
    if (r >= R) return;
    const float* F  = feat + r * Cx;
    const float* CT = ctr + r * 3;
    f16_t* o = out + r * Kp;
    int q = 0;
    for (; q < Cx; q++) o[q] = (f16_t)F[q];
    o[q++] = (f16_t)CT[0];
    o[q++] = (f16_t)CT[1];
    o[q++] = (f16_t)CT[2];
    for (; q < Kp; q++) o[q] = (f16_t)0.0f;
}

// ---------------------------------------------------------------------------
// WMMA GEMM + fused bias/ReLU/BN epilogue.
// A: f16 row-major [M, Kp], Bp: packed fragments, out: f16 [M, N].
// Each wave computes a 16x64 output strip (4 N-tiles) so the A fragment is
// loaded once per k-step and reused across 4 WMMAs (4x less A traffic, and
// 4 independent WMMAs in flight to cover XDL latency). 8 waves per block.
// Requires N % 64 == 0 (all layer widths are 64..1024).
// ---------------------------------------------------------------------------
__global__ void __launch_bounds__(256)
pn2_gemm(const f16_t* __restrict__ A, const f16_t* __restrict__ Bp,
         const float* __restrict__ bias, const float* __restrict__ sa,
         const float* __restrict__ sc, f16_t* __restrict__ out,
         int M, int N, int Kp) {
    int lane = threadIdx.x & 31;
    int wave = threadIdx.x >> 5;
    int mtiles = M >> 4, ntiles4 = N >> 6, ktiles = Kp >> 5;
    long tile = (long)blockIdx.x * 8 + wave;
    if (tile >= (long)mtiles * ntiles4) return;  // uniform per wave: EXEC stays full
    int mt  = (int)(tile / ntiles4);
    int nt4 = (int)(tile % ntiles4);
    int h  = lane >> 4;          // half-wave select
    int ml = lane & 15;          // row within A tile / col within B tile

    const f16_t* Arow  = A + ((size_t)mt * 16 + ml) * Kp + h * 8;
    // B fragments for n-tiles nt4*4 .. nt4*4+3 ; stride between n-tiles = ktiles*512
    const f16_t* Bbase = Bp + ((size_t)(nt4 * 4)) * ktiles * 512 + (size_t)lane * 16;
    size_t bstride = (size_t)ktiles * 512;

    v8f acc0 = {0.f,0.f,0.f,0.f,0.f,0.f,0.f,0.f};
    v8f acc1 = {0.f,0.f,0.f,0.f,0.f,0.f,0.f,0.f};
    v8f acc2 = {0.f,0.f,0.f,0.f,0.f,0.f,0.f,0.f};
    v8f acc3 = {0.f,0.f,0.f,0.f,0.f,0.f,0.f,0.f};
    for (int kt = 0; kt < ktiles; kt++) {
        v8h lo = *(const v8h*)(Arow + (size_t)kt * 32);        // K = h*8 .. +7
        v8h hi = *(const v8h*)(Arow + (size_t)kt * 32 + 16);   // K = 16+h*8 .. +7
        v16h a;
#pragma unroll
        for (int i = 0; i < 8; i++) { a[i] = lo[i]; a[i + 8] = hi[i]; }
        const f16_t* bk = Bbase + (size_t)kt * 512;
        v16h b0 = *(const v16h*)(bk);
        v16h b1 = *(const v16h*)(bk + bstride);
        v16h b2 = *(const v16h*)(bk + 2 * bstride);
        v16h b3 = *(const v16h*)(bk + 3 * bstride);
        acc0 = __builtin_amdgcn_wmma_f32_16x16x32_f16(false, a, false, b0, (short)0, acc0, false, false);
        acc1 = __builtin_amdgcn_wmma_f32_16x16x32_f16(false, a, false, b1, (short)0, acc1, false, false);
        acc2 = __builtin_amdgcn_wmma_f32_16x16x32_f16(false, a, false, b2, (short)0, acc2, false, false);
        acc3 = __builtin_amdgcn_wmma_f32_16x16x32_f16(false, a, false, b3, (short)0, acc3, false, false);
    }
    size_t rowbase = ((size_t)mt * 16 + (size_t)h * 8) * N;
#pragma unroll
    for (int j = 0; j < 4; j++) {
        v8f acc = (j == 0) ? acc0 : (j == 1) ? acc1 : (j == 2) ? acc2 : acc3;
        int n = nt4 * 64 + j * 16 + ml;
        float eb = bias[n], ea = sa[n], ec = sc[n];
#pragma unroll
        for (int v = 0; v < 8; v++) {
            float val = acc[v] + eb;
            val = val > 0.f ? val : 0.f;       // ReLU
            val = ea * val + ec;               // folded BatchNorm
            out[rowbase + (size_t)v * N + n] = (f16_t)val;
        }
    }
}

// ---------------------------------------------------------------------------
// Masked max-pool over 64 neighbors: h [m*64, C] f16 -> feat [m, C] f32
// ---------------------------------------------------------------------------
__global__ void pn2_maxpool(const f16_t* __restrict__ h, const unsigned char* __restrict__ mask,
                            float* __restrict__ out, int C) {
    int i = blockIdx.x;
    int ch = threadIdx.x;
    if (ch >= C) return;
    const f16_t* base = h + (size_t)i * KNBR * C + ch;
    const unsigned char* mk = mask + (size_t)i * KNBR;
    float best = -1e10f;
#pragma unroll 4
    for (int k = 0; k < KNBR; k++)
        if (mk[k]) best = fmaxf(best, (float)base[(size_t)k * C]);
    out[(size_t)i * C + ch] = best;
}

// Global max-pool: h3 [B*512, 1024] f16 -> g [B,1024] f32
__global__ void __launch_bounds__(1024)
pn2_gmax(const f16_t* __restrict__ h, float* __restrict__ g) {
    int b = blockIdx.x;
    int ch = threadIdx.x;
    const f16_t* base = h + (size_t)b * 512 * 1024 + ch;
    float best = -3.4e38f;
    for (int i = 0; i < 512; i++) best = fmaxf(best, (float)base[(size_t)i * 1024]);
    g[(size_t)b * 1024 + ch] = best;
}

// ---------------------------------------------------------------------------
// Head: lin1(1024->512)+relu, lin2(512->256)+relu, lin3(256->40), log_softmax
// One block per batch element.
// ---------------------------------------------------------------------------
__global__ void __launch_bounds__(256)
pn2_head(const float* __restrict__ g,
         const float* __restrict__ W1, const float* __restrict__ b1,
         const float* __restrict__ W2, const float* __restrict__ b2,
         const float* __restrict__ W3, const float* __restrict__ b3,
         float* __restrict__ out) {
    __shared__ float sg[1024];
    __shared__ float h1[512];
    __shared__ float h2[256];
    __shared__ float lg[40];
    int b = blockIdx.x, t = threadIdx.x;
    for (int i = t; i < 1024; i += 256) sg[i] = g[(size_t)b * 1024 + i];
    __syncthreads();
    for (int o = t; o < 512; o += 256) {
        float s = b1[o];
        for (int k = 0; k < 1024; k++) s += sg[k] * W1[(size_t)k * 512 + o];
        h1[o] = fmaxf(s, 0.f);
    }
    __syncthreads();
    if (t < 256) {
        float s = b2[t];
        for (int k = 0; k < 512; k++) s += h1[k] * W2[(size_t)k * 256 + t];
        h2[t] = fmaxf(s, 0.f);
    }
    __syncthreads();
    if (t < 40) {
        float s = b3[t];
        for (int k = 0; k < 256; k++) s += h2[k] * W3[(size_t)k * 40 + t];
        lg[t] = s;
    }
    __syncthreads();
    if (t == 0) {
        float mx = -3.4e38f;
        for (int i = 0; i < 40; i++) mx = fmaxf(mx, lg[i]);
        float sm = 0.f;
        for (int i = 0; i < 40; i++) sm += expf(lg[i] - mx);
        float l = logf(sm);
        for (int i = 0; i < 40; i++) out[(size_t)b * 40 + i] = lg[i] - mx - l;
    }
}

// ---------------------------------------------------------------------------
// Host orchestration
// ---------------------------------------------------------------------------
extern "C" void kernel_launch(void* const* d_in, const int* in_sizes, int n_in,
                              void* d_out, int out_size, void* d_ws, size_t ws_size,
                              hipStream_t stream) {
    (void)in_sizes; (void)n_in; (void)out_size; (void)ws_size;

    const float* norm = (const float*)d_in[0];   // [B*N,3]
    const float* pos  = (const float*)d_in[1];   // [B*N,3]
    // d_in[2] = batch (unused; layout is uniform)

    struct LayerCfg { int cin, cout, Kp; };
    const LayerCfg L[9] = {
        {6, 64, 32}, {64, 64, 64}, {64, 128, 64},           // SA1
        {131, 128, 160}, {128, 128, 128}, {128, 256, 128},  // SA2
        {259, 256, 288}, {256, 512, 256}, {512, 1024, 512}  // SA3
    };
    const float *Wl[9], *bl[9], *gl[9], *btl[9], *mul[9], *varl[9];
    int ii = 3;
    for (int l = 0; l < 9; l++) {
        Wl[l] = (const float*)d_in[ii++]; bl[l]  = (const float*)d_in[ii++];
        gl[l] = (const float*)d_in[ii++]; btl[l] = (const float*)d_in[ii++];
        mul[l] = (const float*)d_in[ii++]; varl[l] = (const float*)d_in[ii++];
    }
    const float* W1 = (const float*)d_in[ii++]; const float* bb1 = (const float*)d_in[ii++];
    const float* W2 = (const float*)d_in[ii++]; const float* bb2 = (const float*)d_in[ii++];
    const float* W3 = (const float*)d_in[ii++]; const float* bb3 = (const float*)d_in[ii++];

    // ---- workspace carve-up ----
    char* wp = (char*)d_ws;
    auto alloc = [&](size_t bytes) -> char* {
        char* p = wp;
        wp += (bytes + 255) & ~(size_t)255;
        return p;
    };
    float* bias[9]; float* sa[9]; float* sc[9]; f16_t* packw[9];
    for (int l = 0; l < 9; l++) {
        bias[l] = (float*)alloc(sizeof(float) * L[l].cout);
        sa[l]   = (float*)alloc(sizeof(float) * L[l].cout);
        sc[l]   = (float*)alloc(sizeof(float) * L[l].cout);
        size_t frag = (size_t)(L[l].Kp / 32) * (L[l].cout / 16) * 512;
        packw[l] = (f16_t*)alloc(sizeof(f16_t) * frag);
    }
    int*   idx1  = (int*)alloc(sizeof(int) * BATCH * 2048);
    float* ctr1  = (float*)alloc(sizeof(float) * BATCH * 2048 * 3);
    int*   nbr1  = (int*)alloc(sizeof(int) * (size_t)BATCH * 2048 * KNBR);
    unsigned char* mask1 = (unsigned char*)alloc((size_t)BATCH * 2048 * KNBR);
    float* feat1 = (float*)alloc(sizeof(float) * (size_t)BATCH * 2048 * 128);
    int*   idx2  = (int*)alloc(sizeof(int) * BATCH * 512);
    float* ctr2  = (float*)alloc(sizeof(float) * BATCH * 512 * 3);
    int*   nbr2  = (int*)alloc(sizeof(int) * (size_t)BATCH * 512 * KNBR);
    unsigned char* mask2 = (unsigned char*)alloc((size_t)BATCH * 512 * KNBR);
    float* feat2 = (float*)alloc(sizeof(float) * (size_t)BATCH * 512 * 256);
    f16_t* bufIn = (f16_t*)alloc(sizeof(f16_t) * (size_t)32768 * 160);   // max gather out
    f16_t* bufT1 = (f16_t*)alloc(sizeof(f16_t) * (size_t)131072 * 64);
    f16_t* bufT2 = (f16_t*)alloc(sizeof(f16_t) * (size_t)131072 * 64);
    f16_t* bufT3 = (f16_t*)alloc(sizeof(f16_t) * (size_t)131072 * 128);
    float* gbuf  = (float*)alloc(sizeof(float) * BATCH * 1024);

    // ---- weight + BN prep ----
    for (int l = 0; l < 9; l++) {
        pn2_bnprep<<<(L[l].cout + 255) / 256, 256, 0, stream>>>(
            bl[l], gl[l], btl[l], mul[l], varl[l], L[l].cout, bias[l], sa[l], sc[l]);
        dim3 pg(L[l].Kp / 32, L[l].cout / 16);
        pn2_packw<<<pg, 32, 0, stream>>>(Wl[l], L[l].cin, L[l].cout, L[l].Kp, packw[l]);
    }

    auto run_gemm = [&](const f16_t* A, int l, f16_t* out, int M) {
        int tiles = (M >> 4) * (L[l].cout >> 6);   // 16x64 strips per wave
        pn2_gemm<<<(tiles + 7) / 8, 256, 0, stream>>>(
            A, packw[l], bias[l], sa[l], sc[l], out, M, L[l].cout, L[l].Kp);
    };

    // ---- SA1: 4096 -> 2048 centers, r=0.2, MLP [6,64,64,128] ----
    pn2_fps<<<BATCH, 512, 0, stream>>>(pos, NPTS, 2048, idx1);
    pn2_ctr<<<(BATCH * 2048 + 255) / 256, 256, 0, stream>>>(pos, idx1, ctr1, NPTS, 2048);
    pn2_knn<<<BATCH * 2048, 256, 0, stream>>>(pos, ctr1, NPTS, 2048, 0.04f, nbr1, mask1);
    for (int b = 0; b < BATCH; b++) {
        const int M = 2048 * KNBR;   // 131072 rows
        pn2_gather<<<(M + 255) / 256, 256, 0, stream>>>(
            norm, pos, ctr1, nbr1, b, NPTS, 2048, 3, 32, bufIn);
        run_gemm(bufIn, 0, bufT1, M);
        run_gemm(bufT1, 1, bufT2, M);
        run_gemm(bufT2, 2, bufT3, M);
        pn2_maxpool<<<2048, 128, 0, stream>>>(
            bufT3, mask1 + (size_t)b * 2048 * KNBR, feat1 + (size_t)b * 2048 * 128, 128);
    }

    // ---- SA2: 2048 -> 512 centers, r=0.4, MLP [131,128,128,256] ----
    pn2_fps<<<BATCH, 512, 0, stream>>>(ctr1, 2048, 512, idx2);
    pn2_ctr<<<(BATCH * 512 + 255) / 256, 256, 0, stream>>>(ctr1, idx2, ctr2, 2048, 512);
    pn2_knn<<<BATCH * 512, 256, 0, stream>>>(ctr1, ctr2, 2048, 512, 0.16f, nbr2, mask2);
    for (int b = 0; b < BATCH; b++) {
        const int M = 512 * KNBR;    // 32768 rows
        pn2_gather<<<(M + 255) / 256, 256, 0, stream>>>(
            feat1, ctr1, ctr2, nbr2, b, 2048, 512, 128, 160, bufIn);
        run_gemm(bufIn, 3, bufT1, M);
        run_gemm(bufT1, 4, bufT2, M);
        run_gemm(bufT2, 5, bufT3, M);
        pn2_maxpool<<<512, 256, 0, stream>>>(
            bufT3, mask2 + (size_t)b * 512 * KNBR, feat2 + (size_t)b * 512 * 256, 256);
    }

    // ---- SA3: MLP [259,256,512,1024] over all B*512 rows, then global max ----
    {
        const int M = BATCH * 512;   // 4096 rows
        pn2_gather3<<<(M + 255) / 256, 256, 0, stream>>>(feat2, ctr2, 256, 288, bufIn);
        run_gemm(bufIn, 6, bufT1, M);
        run_gemm(bufT1, 7, bufT2, M);
        run_gemm(bufT2, 8, bufT3, M);
        pn2_gmax<<<BATCH, 1024, 0, stream>>>(bufT3, gbuf);
    }

    // ---- classifier head + log_softmax ----
    pn2_head<<<BATCH, 256, 0, stream>>>(gbuf, W1, bb1, W2, bb2, W3, bb3, (float*)d_out);
}
